// GridTDModelBU_40389872451960
// MI455X (gfx1250) — compile-verified, matrix operand order checked by reading, not run on hardware
//
#include <hip/hip_runtime.h>
#include <hip/hip_bf16.h>

// MI455X (gfx1250) implementation. All dense GEMMs run on the matrix core via
// v_wmma_f32_16x16x32_bf16 (bf16 inputs, fp32 accumulate). Weights are
// transposed+converted to bf16 once per call into d_ws; the full bf16 weight
// set (~95 MB) is L2-resident (192 MB), so the workload is matrix-core bound,
// not HBM bound. GEMM blocks are 8 waves (M=128); the shared B panel is
// streamed into LDS with global_load_async_to_lds_b128 (ASYNCcnt) and
// double-buffered so WMMA issue overlaps the async copies.

typedef __bf16 bf16_t;
typedef __bf16 v16bf __attribute__((ext_vector_type(16)));
typedef float  v8f   __attribute__((ext_vector_type(8)));

namespace {
constexpr int kB   = 128;
constexpr int kL   = 21;
constexpr int kT   = 20;     // caption_lengths - 1
constexpr int kE   = 512;
constexpr int kH   = 1024;
constexpr int kV   = 12000;
constexpr int kP   = 36;
constexpr int kRaw = 2048;
constexpr int kBP  = kB * kP;          // 4608 rows of Vfeat
constexpr int kN1  = 5 * kH;           // 5120: 4H lstm1 gates + H sentinel gate
constexpr int kN2  = 4 * kH;           // 4096: lstm2 gates
constexpr int kVpad = 12032;           // kV rounded up to 64
constexpr int kKC  = 128;              // K-chunk staged in LDS per buffer
constexpr int kLdsRow = kKC + 8;       // +16B pad -> conflict-free ds reads
}

#define ASYNC_WAIT_4() asm volatile("s_wait_asynccnt 0x4" ::: "memory")
#define ASYNC_WAIT_0() asm volatile("s_wait_asynccnt 0x0" ::: "memory")

// ---------------------------------------------------------------------------
// Fragment load (global or LDS): 16 bf16 = two 16B chunks at +0 and +32B,
// matching the CDNA5 16-bit A 16x32 / B 32x16 per-lane striping (ISA 7.12.2).
// ---------------------------------------------------------------------------
__device__ __forceinline__ v16bf load_frag16(const bf16_t* p) {
  union { uint4 u[2]; v16bf v; } f;
  f.u[0] = *reinterpret_cast<const uint4*>(p);        // K+0..7   (this half)
  f.u[1] = *reinterpret_cast<const uint4*>(p + 16);   // K+16..23 (this half)
  return f.v;
}

// Issue one 64(col) x 128(K) bf16 B-panel chunk: 256 threads x 4 async b128.
__device__ __forceinline__ void issue_panel(const bf16_t* __restrict__ BT, int ldb,
                                            int n0, int kc, bf16_t* buf, int tid) {
  unsigned base = (unsigned)(size_t)buf;   // low 32 bits of flat shared = LDS offset
#pragma unroll
  for (int j = 0; j < 4; ++j) {
    int id  = tid + j * 256;               // 0..1023
    int row = id >> 4;                     // 64 rows
    int g8  = id & 15;                     // 16 groups of 8 bf16 per row
    const bf16_t* g = BT + (long long)(n0 + row) * ldb + kc + g8 * 8;
    unsigned l = base + (unsigned)((row * kLdsRow + g8 * 8) * 2);
    asm volatile("global_load_async_to_lds_b128 %0, %1, off"
                 :: "v"(l), "v"(g) : "memory");
  }
}

// One K-chain of the block GEMM: stream B panel chunks into LDS (double
// buffered, async), multiply against this wave's 16 A rows.
__device__ __forceinline__ void gemm_chain(const bf16_t* __restrict__ Ac, int ldac,
                                           const bf16_t* __restrict__ Bc, int Kc,
                                           int m0, int n0, int lrow, int half,
                                           int tid, bf16_t* sbuf /* [2][64*kLdsRow] */,
                                           v8f acc[4]) {
  const int nchunk = Kc / kKC;
  issue_panel(Bc, Kc, n0, 0, sbuf, tid);
  for (int ch = 0; ch < nchunk; ++ch) {
    bf16_t* cur = sbuf + (ch & 1) * (64 * kLdsRow);
    if (ch + 1 < nchunk) {
      issue_panel(Bc, Kc, n0, (ch + 1) * kKC, sbuf + ((ch & 1) ^ 1) * (64 * kLdsRow), tid);
      ASYNC_WAIT_4();            // chunk ch complete; chunk ch+1 in flight
    } else {
      ASYNC_WAIT_0();
    }
    __syncthreads();             // chunk ch visible to all waves

    const bf16_t* arow = Ac + (long long)(m0 + lrow) * ldac + ch * kKC + half * 8;
    // Preload the whole chunk's A fragments (global, one clause / one wait).
    v16bf afr[4];
#pragma unroll
    for (int g = 0; g < 4; ++g) afr[g] = load_frag16(arow + g * 32);
#pragma unroll
    for (int g = 0; g < 4; ++g) {
      // Load all 4 B fragments first so the ds_loads clause together and the
      // 4 WMMAs issue back-to-back behind a single dscnt wait.
      v16bf b0 = load_frag16(&cur[(0 * 16 + lrow) * kLdsRow + g * 32 + half * 8]);
      v16bf b1 = load_frag16(&cur[(1 * 16 + lrow) * kLdsRow + g * 32 + half * 8]);
      v16bf b2 = load_frag16(&cur[(2 * 16 + lrow) * kLdsRow + g * 32 + half * 8]);
      v16bf b3 = load_frag16(&cur[(3 * 16 + lrow) * kLdsRow + g * 32 + half * 8]);
      acc[0] = __builtin_amdgcn_wmma_f32_16x16x32_bf16(false, afr[g], false, b0,
                                                       (short)0, acc[0], false, false);
      acc[1] = __builtin_amdgcn_wmma_f32_16x16x32_bf16(false, afr[g], false, b1,
                                                       (short)0, acc[1], false, false);
      acc[2] = __builtin_amdgcn_wmma_f32_16x16x32_bf16(false, afr[g], false, b2,
                                                       (short)0, acc[2], false, false);
      acc[3] = __builtin_amdgcn_wmma_f32_16x16x32_bf16(false, afr[g], false, b3,
                                                       (short)0, acc[3], false, false);
    }
    __syncthreads();             // everyone done reading cur before reuse
  }
}

// ---------------------------------------------------------------------------
// WMMA GEMM: C[M x N] = A(bf16, MxK) * BT(bf16, Npad x K) [+ A2*BT2 chain]
// + bias, optional relu, optional bf16 mirror. Block = 256 threads = 8 waves;
// wave w owns rows [16w,16w+16) of a 128-row stripe; 64 output columns/block.
// B panel is shared by all 8 waves via async-to-LDS double buffering.
// Requires: M % 128 == 0, K % 128 == 0 (and K2 % 128 == 0 if used).
// ---------------------------------------------------------------------------
__global__ void __launch_bounds__(256)
k_wmma_gemm(const bf16_t* __restrict__ A, int lda,
            const bf16_t* __restrict__ BT,
            const bf16_t* __restrict__ A2, int lda2,
            const bf16_t* __restrict__ BT2, int K2,
            const float* __restrict__ bias,
            float* __restrict__ C, long long ldc,
            bf16_t* __restrict__ Cb, int ldcb,
            int N, int K, int relu) {
  __shared__ bf16_t sb[2][64 * kLdsRow];

  const int tid  = threadIdx.x;
  const int lane = tid & 31;
  const int wave = tid >> 5;
  const int half = lane >> 4;     // which K-half of the 32-chunk this lane holds
  const int lrow = lane & 15;     // A row / B column within the 16-tile
  const int m0 = blockIdx.y * 128 + wave * 16;
  const int n0 = blockIdx.x * 64;

  v8f acc[4] = {};

  gemm_chain(A, lda, BT, K, m0, n0, lrow, half, tid, &sb[0][0], acc);
  if (A2 != nullptr)
    gemm_chain(A2, lda2, BT2, K2, m0, n0, lrow, half, tid, &sb[0][0], acc);

#pragma unroll
  for (int tl = 0; tl < 4; ++tl) {
    int col = n0 + tl * 16 + lrow;
    if (col >= N) continue;
    float bb = bias ? bias[col] : 0.0f;
#pragma unroll
    for (int r = 0; r < 8; ++r) {
      int row = m0 + r + half * 8;        // C/D layout: VGPR r -> M = r + 8*half
      float v = acc[tl][r] + bb;
      if (relu) v = fmaxf(v, 0.0f);
      C[(long long)row * ldc + col] = v;
      if (Cb) Cb[(long long)row * ldcb + col] = (bf16_t)v;
    }
  }
}

// ---------------------------------------------------------------------------
// Pre-pass helpers
// ---------------------------------------------------------------------------
__global__ void k_transpose_bf16(const float* __restrict__ W, bf16_t* __restrict__ WT,
                                 int K, int N, int Npad) {
  long long i = (long long)blockIdx.x * blockDim.x + threadIdx.x;
  long long total = (long long)Npad * K;
  if (i >= total) return;
  int n = (int)(i / K), k = (int)(i % K);
  WT[i] = (bf16_t)((n < N) ? W[(long long)k * N + n] : 0.0f);
}

// Transposed concat of two weight matrices sharing K: rows [0,Na) from Wa,
// rows [Na, Na+Nb) from Wb. Used to fuse LSTM1 gates + sentinel gate.
__global__ void k_transpose_cat_bf16(const float* __restrict__ Wa, int Na,
                                     const float* __restrict__ Wb, int Nb,
                                     bf16_t* __restrict__ WT, int K) {
  long long i = (long long)blockIdx.x * blockDim.x + threadIdx.x;
  long long total = (long long)(Na + Nb) * K;
  if (i >= total) return;
  int n = (int)(i / K), k = (int)(i % K);
  WT[i] = (bf16_t)((n < Na) ? Wa[(long long)k * Na + n]
                            : Wb[(long long)k * Nb + (n - Na)]);
}

__global__ void k_vec_add(const float* a, const float* b, float* o, int n) {
  int i = blockIdx.x * blockDim.x + threadIdx.x;
  if (i < n) o[i] = a[i] + b[i];
}

__global__ void k_f32_to_bf16(const float* __restrict__ s, bf16_t* __restrict__ d, long long n) {
  long long i = (long long)blockIdx.x * blockDim.x + threadIdx.x;
  if (i < n) d[i] = (bf16_t)s[i];
}

__global__ void k_zero_u32(unsigned* p, long long n) {
  long long i = (long long)blockIdx.x * blockDim.x + threadIdx.x;
  if (i < n) p[i] = 0u;
}

__global__ void k_mean_feat(const float* __restrict__ Vfeat, bf16_t* __restrict__ o) {
  int i = blockIdx.x * blockDim.x + threadIdx.x;
  if (i >= kB * kH) return;
  int b = i / kH, h = i % kH;
  float s = 0.0f;
  for (int p = 0; p < kP; ++p) s += Vfeat[((long long)b * kP + p) * kH + h];
  o[i] = (bf16_t)(s * (1.0f / kP));
}

__global__ void k_embed_gather(const int* __restrict__ caps, const float* __restrict__ emb,
                               bf16_t* __restrict__ we) {
  long long i = (long long)blockIdx.x * blockDim.x + threadIdx.x;
  if (i >= (long long)kB * kT * kE) return;
  int b = (int)(i / (kT * kE));
  int r = (int)(i % (kT * kE));
  int t = r / kE, e = r % kE;
  int idx = caps[b * kL + t];
  we[i] = (bf16_t)emb[(long long)idx * kE + e];
}

// ---------------------------------------------------------------------------
// Per-step elementwise kernels
// ---------------------------------------------------------------------------
__device__ __forceinline__ float sigf(float x) { return 1.0f / (1.0f + expf(-x)); }

__global__ void k_build_xt(const bf16_t* __restrict__ h2b, const bf16_t* __restrict__ globb,
                           const bf16_t* __restrict__ we, int t, bf16_t* __restrict__ xt) {
  int i = blockIdx.x * blockDim.x + threadIdx.x;
  if (i >= kB * (kH + 2 * kE)) return;
  int b = i / (kH + 2 * kE), j = i % (kH + 2 * kE);
  bf16_t v;
  if (j < kH)            v = h2b[b * kH + j];
  else if (j < kH + kE)  v = globb[b * kE + (j - kH)];
  else                   v = we[((long long)b * kT + t) * kE + (j - kH - kE)];
  xt[i] = v;
}

// g layout per row: [i(H) f(H) g(H) o(H) sen(H)], ld = kN1
__global__ void k_lstm1_pw(const float* __restrict__ g, float* __restrict__ c1,
                           bf16_t* __restrict__ h1b, bf16_t* __restrict__ xcat2,
                           float* __restrict__ st) {
  int i = blockIdx.x * blockDim.x + threadIdx.x;
  if (i >= kB * kH) return;
  int b = i / kH, h = i % kH;
  const float* gb = g + (long long)b * kN1;
  float ig = sigf(gb[h]);
  float fg = sigf(gb[kH + h]);
  float gg = tanhf(gb[2 * kH + h]);
  float og = sigf(gb[3 * kH + h]);
  float c  = fg * c1[i] + ig * gg;
  float hn = og * tanhf(c);
  c1[i] = c;
  h1b[i] = (bf16_t)hn;
  xcat2[b * (2 * kH) + kH + h] = (bf16_t)hn;   // second half of lstm2 input
  float sp = sigf(gb[4 * kH + h]);             // sentinel gate pre-act
  st[i] = sp * tanhf(c);
}

// One workgroup per batch element: ht_proj, att scores, softmaxes, context, c_hat.
__global__ void k_attention(const bf16_t* __restrict__ h1b, const float* __restrict__ st,
                            const float* __restrict__ Wg, const float* __restrict__ Ws,
                            const float* __restrict__ bs, const float* __restrict__ wh,
                            const float* __restrict__ img_proj, const float* __restrict__ Vfeat,
                            int t, float* __restrict__ out_alpha, float* __restrict__ out_beta,
                            float* __restrict__ c_hat, bf16_t* __restrict__ xcat2) {
  __shared__ float ht[kP], sws[kP], zz[kP + 1], al[kP], sm[3];
  int b = blockIdx.x, tid = threadIdx.x;
  if (tid < kP) {                       // ht_proj[b, p] = h1n . Wg[:, p]
    float s = 0.0f;
    for (int h = 0; h < kH; ++h) s += (float)h1b[b * kH + h] * Wg[h * kP + tid];
    ht[tid] = s;
  } else if (tid < 2 * kP) {            // st @ Ws + bs
    int p = tid - kP;
    float s = bs[p];
    for (int h = 0; h < kH; ++h) s += st[b * kH + h] * Ws[h * kP + p];
    sws[p] = s;
  }
  __syncthreads();
  if (tid < kP) {                        // z[b,p] = sum_j tanh(img_proj + ht[p]) wh[j]
    float hp = ht[tid], s = 0.0f;
    const float* ip = img_proj + ((long long)b * kP + tid) * kP;
    for (int j = 0; j < kP; ++j) s += tanhf(ip[j] + hp) * wh[j];
    zz[tid] = s;
  }
  if (tid == kP) {                       // att_vs[b] = sum_j tanh(sws[j]+ht[j]) wh[j]
    float s = 0.0f;
    for (int j = 0; j < kP; ++j) s += tanhf(sws[j] + ht[j]) * wh[j];
    zz[kP] = s;
  }
  __syncthreads();
  if (tid == 0) {
    float m = -1e30f;
    for (int p = 0; p < kP; ++p) m = fmaxf(m, zz[p]);
    float d = 0.0f;
    for (int p = 0; p < kP; ++p) d += expf(zz[p] - m);
    float m2 = fmaxf(m, zz[kP]);
    float d2 = 0.0f;
    for (int p = 0; p <= kP; ++p) d2 += expf(zz[p] - m2);
    sm[0] = m; sm[1] = d; sm[2] = expf(zz[kP] - m2) / d2;   // beta
  }
  __syncthreads();
  float beta = sm[2];
  if (tid < kP) {
    float a = expf(zz[tid] - sm[0]) / sm[1];
    al[tid] = a;
    out_alpha[((long long)b * kT + t) * kP + tid] = a;
  }
  if (tid == 0) out_beta[b * kT + t] = beta;
  __syncthreads();
  for (int h = tid; h < kH; h += blockDim.x) {
    float ctx = 0.0f;
    for (int p = 0; p < kP; ++p) ctx += al[p] * Vfeat[((long long)b * kP + p) * kH + h];
    float ch = beta * st[b * kH + h] + (1.0f - beta) * ctx;
    c_hat[b * kH + h] = ch;
    xcat2[b * (2 * kH) + h] = (bf16_t)ch;    // first half of lstm2 input
  }
}

__global__ void k_lstm2_pw(const float* __restrict__ g, float* __restrict__ c2,
                           bf16_t* __restrict__ h2b, const float* __restrict__ c_hat,
                           bf16_t* __restrict__ fcin) {
  int i = blockIdx.x * blockDim.x + threadIdx.x;
  if (i >= kB * kH) return;
  int b = i / kH, h = i % kH;
  const float* gb = g + (long long)b * kN2;
  float ig = sigf(gb[h]);
  float fg = sigf(gb[kH + h]);
  float gg = tanhf(gb[2 * kH + h]);
  float og = sigf(gb[3 * kH + h]);
  float c  = fg * c2[i] + ig * gg;
  float hn = og * tanhf(c);
  c2[i] = c;
  h2b[i] = (bf16_t)hn;
  fcin[i] = (bf16_t)(c_hat[i] + hn);          // score input = c_hat + h2n
}

// ---------------------------------------------------------------------------
// Host launcher
// ---------------------------------------------------------------------------
static inline int cdiv(long long a, long long b) { return (int)((a + b - 1) / b); }

static void launch_gemm(hipStream_t s, const bf16_t* A, int lda, const bf16_t* BT,
                        const bf16_t* A2, int lda2, const bf16_t* BT2, int K2,
                        const float* bias, float* C, long long ldc,
                        bf16_t* Cb, int ldcb, int M, int N, int K, int relu) {
  dim3 grid(cdiv(N, 64), M / 128);
  k_wmma_gemm<<<grid, dim3(256), 0, s>>>(A, lda, BT, A2, lda2, BT2, K2,
                                         bias, C, ldc, Cb, ldcb, N, K, relu);
}

extern "C" void kernel_launch(void* const* d_in, const int* in_sizes, int n_in,
                              void* d_out, int out_size, void* d_ws, size_t ws_size,
                              hipStream_t stream) {
  const float* images  = (const float*)d_in[0];
  const int*   caps    = (const int*)d_in[1];
  // d_in[2] = caption_lengths scalar; trace-time constant L=21 -> T=20
  const float* W_img   = (const float*)d_in[3];
  const float* b_img   = (const float*)d_in[4];
  const float* W_glob  = (const float*)d_in[5];
  const float* b_glob  = (const float*)d_in[6];
  const float* W_ih1   = (const float*)d_in[7];
  const float* W_hh1   = (const float*)d_in[8];
  const float* b_ih1   = (const float*)d_in[9];
  const float* b_hh1   = (const float*)d_in[10];
  const float* Wx_gate = (const float*)d_in[11];
  const float* bx_gate = (const float*)d_in[12];
  const float* Wh_gate = (const float*)d_in[13];
  const float* bh_gate = (const float*)d_in[14];
  const float* W_ih2   = (const float*)d_in[15];
  const float* W_hh2   = (const float*)d_in[16];
  const float* b_ih2   = (const float*)d_in[17];
  const float* b_hh2   = (const float*)d_in[18];
  const float* Wv      = (const float*)d_in[19];
  const float* bv      = (const float*)d_in[20];
  const float* Ws      = (const float*)d_in[21];
  const float* bs      = (const float*)d_in[22];
  const float* Wg      = (const float*)d_in[23];
  const float* wh      = (const float*)d_in[24];
  const float* emb     = (const float*)d_in[25];
  const float* W_fc    = (const float*)d_in[26];
  const float* b_fc    = (const float*)d_in[27];

  float* out_preds  = (float*)d_out;
  float* out_alphas = out_preds + (long long)kB * kT * kV;
  float* out_betas  = out_alphas + (long long)kB * kT * kP;

  // ---- workspace bump allocator (256B aligned) ----
  size_t off = 0;
  auto alloc = [&](size_t bytes) -> void* {
    void* p = (char*)d_ws + off;
    off += (bytes + 255) & ~(size_t)255;
    return p;
  };
  bf16_t* WTcx  = (bf16_t*)alloc((size_t)kN1 * kRaw * 2);   // [5120 x 2048] ih1|Wx
  bf16_t* WTch  = (bf16_t*)alloc((size_t)kN1 * kH * 2);     // [5120 x 1024] hh1|Wh
  float*  bias1 = (float*) alloc((size_t)kN1 * 4);
  bf16_t* WTih2 = (bf16_t*)alloc((size_t)kN2 * (2 * kH) * 2);
  bf16_t* WThh2 = (bf16_t*)alloc((size_t)kN2 * kH * 2);
  float*  bias2 = (float*) alloc((size_t)kN2 * 4);
  bf16_t* WTimg = (bf16_t*)alloc((size_t)kH * kRaw * 2);
  bf16_t* WTglb = (bf16_t*)alloc((size_t)kE * kH * 2);
  bf16_t* WTfc  = (bf16_t*)alloc((size_t)kVpad * kH * 2);
  bf16_t* WTv   = (bf16_t*)alloc((size_t)64 * kH * 2);      // Wv^T padded to 64
  bf16_t* imgb  = (bf16_t*)alloc((size_t)kBP * kRaw * 2);
  float*  Vfeat = (float*) alloc((size_t)kBP * kH * 4);
  bf16_t* Vfb   = (bf16_t*)alloc((size_t)kBP * kH * 2);
  float*  imgpr = (float*) alloc((size_t)kBP * kP * 4);
  bf16_t* gmean = (bf16_t*)alloc((size_t)kB * kH * 2);
  float*  globf = (float*) alloc((size_t)kB * kE * 4);
  bf16_t* globb = (bf16_t*)alloc((size_t)kB * kE * 2);
  bf16_t* we    = (bf16_t*)alloc((size_t)kB * kT * kE * 2);
  float*  c1    = (float*) alloc((size_t)kB * kH * 4);
  float*  c2    = (float*) alloc((size_t)kB * kH * 4);
  float*  st    = (float*) alloc((size_t)kB * kH * 4);
  float*  chat  = (float*) alloc((size_t)kB * kH * 4);
  bf16_t* h1b   = (bf16_t*)alloc((size_t)kB * kH * 2);
  bf16_t* h2b   = (bf16_t*)alloc((size_t)kB * kH * 2);
  bf16_t* xt    = (bf16_t*)alloc((size_t)kB * (2 * kH) * 2);
  bf16_t* xcat2 = (bf16_t*)alloc((size_t)kB * (2 * kH) * 2);
  float*  g1    = (float*) alloc((size_t)kB * kN1 * 4);
  float*  g2    = (float*) alloc((size_t)kB * kN2 * 4);
  bf16_t* fcin  = (bf16_t*)alloc((size_t)kB * kH * 2);
  (void)ws_size; (void)in_sizes; (void)n_in; (void)out_size;

  const int TB = 256;

  // ---- pre-pass: weight conversion / transposition (bf16, N x K layout) ----
  { long long n = (long long)kN1 * kRaw;
    k_transpose_cat_bf16<<<cdiv(n, TB), TB, 0, stream>>>(W_ih1, 4 * kH, Wx_gate, kH, WTcx, kRaw); }
  { long long n = (long long)kN1 * kH;
    k_transpose_cat_bf16<<<cdiv(n, TB), TB, 0, stream>>>(W_hh1, 4 * kH, Wh_gate, kH, WTch, kH); }
  k_vec_add<<<cdiv(4 * kH, TB), TB, 0, stream>>>(b_ih1, b_hh1, bias1, 4 * kH);
  k_vec_add<<<cdiv(kH, TB), TB, 0, stream>>>(bx_gate, bh_gate, bias1 + 4 * kH, kH);
  k_vec_add<<<cdiv(kN2, TB), TB, 0, stream>>>(b_ih2, b_hh2, bias2, kN2);
  { long long n = (long long)kN2 * 2 * kH;
    k_transpose_bf16<<<cdiv(n, TB), TB, 0, stream>>>(W_ih2, WTih2, 2 * kH, kN2, kN2); }
  { long long n = (long long)kN2 * kH;
    k_transpose_bf16<<<cdiv(n, TB), TB, 0, stream>>>(W_hh2, WThh2, kH, kN2, kN2); }
  { long long n = (long long)kH * kRaw;
    k_transpose_bf16<<<cdiv(n, TB), TB, 0, stream>>>(W_img, WTimg, kRaw, kH, kH); }
  { long long n = (long long)kE * kH;
    k_transpose_bf16<<<cdiv(n, TB), TB, 0, stream>>>(W_glob, WTglb, kH, kE, kE); }
  { long long n = (long long)kVpad * kH;
    k_transpose_bf16<<<cdiv(n, TB), TB, 0, stream>>>(W_fc, WTfc, kH, kV, kVpad); }
  { long long n = (long long)64 * kH;
    k_transpose_bf16<<<cdiv(n, TB), TB, 0, stream>>>(Wv, WTv, kH, kP, 64); }

  // images -> bf16; state zeroing; embedding gather
  { long long n = (long long)kBP * kRaw;
    k_f32_to_bf16<<<cdiv(n, TB), TB, 0, stream>>>(images, imgb, n); }
  k_zero_u32<<<cdiv(kB * kH, TB), TB, 0, stream>>>((unsigned*)c1, kB * kH);
  k_zero_u32<<<cdiv(kB * kH, TB), TB, 0, stream>>>((unsigned*)c2, kB * kH);
  k_zero_u32<<<cdiv(kB * kH / 2, TB), TB, 0, stream>>>((unsigned*)h1b, kB * kH / 2);
  k_zero_u32<<<cdiv(kB * kH / 2, TB), TB, 0, stream>>>((unsigned*)h2b, kB * kH / 2);
  { long long n = (long long)kB * kT * kE;
    k_embed_gather<<<cdiv(n, TB), TB, 0, stream>>>(caps, emb, we); }

  // Vfeat = relu(img @ W_img + b_img)  (also bf16 mirror)
  launch_gemm(stream, imgb, kRaw, WTimg, nullptr, 0, nullptr, 0,
              b_img, Vfeat, kH, Vfb, kH, kBP, kH, kRaw, 1);
  // glob = relu(mean(Vfeat) @ W_glob + b_glob)
  k_mean_feat<<<cdiv(kB * kH, TB), TB, 0, stream>>>(Vfeat, gmean);
  launch_gemm(stream, gmean, kH, WTglb, nullptr, 0, nullptr, 0,
              b_glob, globf, kE, globb, kE, kB, kE, kH, 1);
  // img_proj = Vfeat @ Wv + bv   (N=36 padded to 64; store guarded)
  launch_gemm(stream, Vfb, kH, WTv, nullptr, 0, nullptr, 0,
              bv, imgpr, kP, nullptr, 0, kBP, kP, kH, 0);

  // ---- time loop ----
  for (int t = 0; t < kT; ++t) {
    k_build_xt<<<cdiv(kB * 2 * kH, TB), TB, 0, stream>>>(h2b, globb, we, t, xt);
    // g1 = xt @ [W_ih1|Wx]^ + h1 @ [W_hh1|Wh]^ + biases  (N=5120, dual-K WMMA)
    launch_gemm(stream, xt, 2 * kH, WTcx, h1b, kH, WTch, kH,
                bias1, g1, kN1, nullptr, 0, kB, kN1, 2 * kH, 0);
    k_lstm1_pw<<<cdiv(kB * kH, TB), TB, 0, stream>>>(g1, c1, h1b, xcat2, st);
    k_attention<<<kB, 256, 0, stream>>>(h1b, st, Wg, Ws, bs, wh, imgpr, Vfeat,
                                        t, out_alphas, out_betas, chat, xcat2);
    // g2 = [c_hat,h1n] @ W_ih2^ + h2 @ W_hh2^ + biases
    launch_gemm(stream, xcat2, 2 * kH, WTih2, h2b, kH, WThh2, kH,
                bias2, g2, kN2, nullptr, 0, kB, kN2, 2 * kH, 0);
    k_lstm2_pw<<<cdiv(kB * kH, TB), TB, 0, stream>>>(g2, c2, h2b, chat, fcin);
    // preds[:, t, :] = (c_hat + h2n) @ W_fc + b_fc   (stored straight into d_out)
    launch_gemm(stream, fcin, kH, WTfc, nullptr, 0, nullptr, 0,
                b_fc, out_preds + (long long)t * kV, (long long)kT * kV,
                nullptr, 0, kB, kV, kH, 0);
  }
}
